// ProgressiveGAT_76605036691788
// MI455X (gfx1250) — matrix operand order repaired and checked by reading, not compiled, not run on hardware
//
#include <hip/hip_runtime.h>

// ---------------------------------------------------------------------------
// ProgressiveGAT (GATv2 x2 -> mean pool -> 2-layer LSTM -> MLP head) for
// MI455X / gfx1250.  Matrix work uses v_wmma_f32_16x16x32_f16 (wave32).
// ---------------------------------------------------------------------------

typedef _Float16 v8h  __attribute__((ext_vector_type(8)));
typedef _Float16 v16h __attribute__((ext_vector_type(16)));
typedef float    v8f  __attribute__((ext_vector_type(8)));

#define BB   16
#define TT   32
#define NN   64
#define EE   512
#define ET   576      // E + N self loops
#define DD   64
#define HH   4
#define LL   2
#define SHD  128
#define GG   (BB*TT)  // 512 graphs

// f16 workspace layout (element offsets)
#define OFF_WL    0          // gat_Wl  : L*256*64 = 32768
#define OFF_WR    32768      // gat_Wr  : 32768
#define OFF_WIH0  65536      // 512*64  = 32768
#define OFF_WHH0  98304      // 512*128 = 65536
#define OFF_WIH1  163840     // 512*128
#define OFF_WHH1  229376     // 512*128
#define OFF_EMB   294912     // G*64    = 32768
#define CVT_TOTAL 294912

// ---------------- WMMA fragment helpers (16x16x32 f16; A/B in contiguous k order) -------
// A (16x32, f16): lane m = lane&15, half = lane>>4;
//   elem i (0..7)  -> K = kc + half*16 + i
//   elem 8+i       -> K = kc + half*16 + 8 + i
__device__ __forceinline__ v16h load_a_frag(const _Float16* base, int ld,
                                            int m0, int kc, int lane) {
  int m = m0 + (lane & 15);
  int half = lane >> 4;
  const v8h lo = *(const v8h*)(base + m * ld + kc + half * 16);
  const v8h hi = *(const v8h*)(base + m * ld + kc + half * 16 + 8);
  v16h a;
#pragma unroll
  for (int i = 0; i < 8; ++i) { a[i] = lo[i]; a[8 + i] = hi[i]; }
  return a;
}

// B (32x16, f16) where B[k][n] = W[n0+n][kc+k] (W row-major, stride ld):
//   lane n = lane&15, half = lane>>4; elem i -> K = kc + half*16 + i
//   -> one contiguous 32-byte load per lane.
__device__ __forceinline__ v16h load_b_frag(const _Float16* W, int ld,
                                            int n0, int kc, int lane) {
  int n = n0 + (lane & 15);
  int half = lane >> 4;
  return *(const v16h*)(W + n * ld + kc + half * 16);
}

// C/D (16x16, f32): lane n = lane&15, half = lane>>4; vgpr r -> M = half*8 + r
__device__ __forceinline__ void store_c_frag(float* base, int ld,
                                             int m0, int n0, int lane, v8f c) {
  int n = n0 + (lane & 15);
  int half = lane >> 4;
#pragma unroll
  for (int r = 0; r < 8; ++r) base[(m0 + half * 8 + r) * ld + n] = c[r];
}

__device__ __forceinline__ v8f wmma_f16(v16h a, v16h b, v8f c) {
  return __builtin_amdgcn_wmma_f32_16x16x32_f16(false, a, false, b,
                                                (short)0, c, false, false);
}

// sortable-uint mapping for float atomic max (handles negatives)
__device__ __forceinline__ unsigned f2sort(float f) {
  unsigned u = __float_as_uint(f);
  return (u & 0x80000000u) ? ~u : (u | 0x80000000u);
}
__device__ __forceinline__ float sort2f(unsigned u) {
  unsigned v = (u & 0x80000000u) ? (u & 0x7FFFFFFFu) : ~u;
  return __uint_as_float(v);
}
__device__ __forceinline__ float sigm(float x) { return 1.f / (1.f + __expf(-x)); }

// ---------------------------------------------------------------------------
// Kernel 1: convert WMMA-consumed weights to f16 in workspace
// ---------------------------------------------------------------------------
__global__ __launch_bounds__(256)
void xg_convert_kernel(const float* __restrict__ wl, const float* __restrict__ wr,
                       const float* __restrict__ wih0, const float* __restrict__ whh0,
                       const float* __restrict__ wih1, const float* __restrict__ whh1,
                       _Float16* __restrict__ ws16) {
  int i = blockIdx.x * 256 + threadIdx.x;
  if (i >= CVT_TOTAL) return;
  if (i < 32768)        ws16[OFF_WL   + i]            = (_Float16)wl[i];
  else if (i < 65536)   ws16[OFF_WR   + (i -  32768)] = (_Float16)wr[i - 32768];
  else if (i < 98304)   ws16[OFF_WIH0 + (i -  65536)] = (_Float16)wih0[i - 65536];
  else if (i < 163840)  ws16[OFF_WHH0 + (i -  98304)] = (_Float16)whh0[i - 98304];
  else if (i < 229376)  ws16[OFF_WIH1 + (i - 163840)] = (_Float16)wih1[i - 163840];
  else                  ws16[OFF_WHH1 + (i - 229376)] = (_Float16)whh1[i - 229376];
}

// ---------------------------------------------------------------------------
// Kernel 2: per-graph projection + 2x GATv2 + layernorm/relu + mean pool
// One workgroup (8 waves) per graph; everything staged in LDS.
// ---------------------------------------------------------------------------
__global__ __launch_bounds__(256)
void xg_gat_kernel(const float* __restrict__ x, const int* __restrict__ ei,
                   const float* __restrict__ projW, const float* __restrict__ projb,
                   const float* __restrict__ gatt, const float* __restrict__ gbias,
                   const float* __restrict__ lng, const float* __restrict__ lnb,
                   _Float16* __restrict__ ws16) {
  __shared__ _Float16 h16[NN * DD];         // node features, f16 (A operand)
  __shared__ float xl_sh[NN * DD];          // per-head source transform
  __shared__ float xr_sh[NN * DD];          // per-head target transform
  __shared__ float oac[NN * DD];            // head-mean accumulator / output
  __shared__ float logit_sh[ET];
  __shared__ unsigned mmax[NN];
  __shared__ float ssum[NN];
  __shared__ int srcs[ET], dsts[ET];
  __shared__ float att_sh[DD];

  const int g    = blockIdx.x;
  const int tid  = threadIdx.x;
  const int lane = tid & 31;
  const int wave = tid >> 5;

  const _Float16* WL16 = ws16 + OFF_WL;
  const _Float16* WR16 = ws16 + OFF_WR;

  // ---- input projection + relu: h = relu(x @ projW.T + projb) ----
  const float* xg = x + g * NN * 2;
  for (int idx = tid; idx < NN * DD; idx += 256) {
    int n = idx >> 6, d = idx & 63;
    float v = xg[n * 2 + 0] * projW[d * 2 + 0] +
              xg[n * 2 + 1] * projW[d * 2 + 1] + projb[d];
    h16[idx] = (_Float16)fmaxf(v, 0.f);
  }
  // ---- edge list with self loops ----
  const int* eg = ei + g * 2 * EE;
  for (int e = tid; e < ET; e += 256) {
    if (e < EE) { srcs[e] = eg[e]; dsts[e] = eg[EE + e]; }
    else        { srcs[e] = e - EE; dsts[e] = e - EE; }
  }
  __syncthreads();

  for (int l = 0; l < LL; ++l) {
    for (int idx = tid; idx < NN * DD; idx += 256) oac[idx] = 0.f;
    __syncthreads();

    for (int hh = 0; hh < HH; ++hh) {
      if (tid < DD) att_sh[tid] = gatt[(l * HH + hh) * DD + tid];
      if (tid < NN) { mmax[tid] = 0u; ssum[tid] = 0.f; }
      __syncthreads();

      // xl = h @ Wl_h.T, xr = h @ Wr_h.T  (each 64x64, 16 tiles) via WMMA
      for (int j = 0; j < 4; ++j) {
        int job = wave * 4 + j;              // 0..31 : 16 xl tiles + 16 xr tiles
        const _Float16* Wg = (job < 16) ? WL16 : WR16;
        float* outp = (job < 16) ? xl_sh : xr_sh;
        int tl = job & 15, tm = tl >> 2, tn = tl & 3;
        const _Float16* Wbase = Wg + (l * 256 + hh * 64) * DD; // row-major [64][64]
        v8f acc = {};
#pragma unroll
        for (int kc = 0; kc < DD; kc += 32) {
          v16h a = load_a_frag(h16, DD, tm * 16, kc, lane);
          v16h b = load_b_frag(Wbase, DD, tn * 16, kc, lane);
          acc = wmma_f16(a, b, acc);
        }
        store_c_frag(outp, DD, tm * 16, tn * 16, lane, acc);
      }
      __syncthreads();

      // pass 1: edge logits + segment max over dst
      for (int e = tid; e < ET; e += 256) {
        int s = srcs[e], dn = dsts[e];
        float lg = 0.f;
        for (int d = 0; d < DD; ++d) {
          float v = xl_sh[s * DD + d] + xr_sh[dn * DD + d];
          v = (v > 0.f) ? v : 0.2f * v;          // leaky_relu(0.2)
          lg += v * att_sh[d];
        }
        logit_sh[e] = lg;
        atomicMax(&mmax[dn], f2sort(lg));
      }
      __syncthreads();
      // pass 2: exp + segment sum
      for (int e = tid; e < ET; e += 256) {
        int dn = dsts[e];
        float a = __expf(logit_sh[e] - sort2f(mmax[dn]));
        logit_sh[e] = a;
        atomicAdd(&ssum[dn], a);
      }
      __syncthreads();
      // pass 3: weighted scatter (fold 1/H head mean)
      for (int e4 = tid; e4 < ET * 4; e4 += 256) {
        int e = e4 >> 2, d0 = (e4 & 3) * 16;
        int s = srcs[e], dn = dsts[e];
        float alpha = logit_sh[e] / ssum[dn] * 0.25f;
        for (int d = d0; d < d0 + 16; ++d)
          atomicAdd(&oac[dn * DD + d], xl_sh[s * DD + d] * alpha);
      }
      __syncthreads();
    }

    // bias + layernorm + relu, refresh f16 features
    if (tid < NN) {
      int n = tid;
      float mu = 0.f;
      for (int d = 0; d < DD; ++d) mu += oac[n * DD + d] + gbias[l * DD + d];
      mu *= (1.f / DD);
      float var = 0.f;
      for (int d = 0; d < DD; ++d) {
        float v = oac[n * DD + d] + gbias[l * DD + d] - mu;
        var += v * v;
      }
      var *= (1.f / DD);
      float inv = rsqrtf(var + 1e-5f);
      for (int d = 0; d < DD; ++d) {
        float v = oac[n * DD + d] + gbias[l * DD + d];
        float y = lng[l * DD + d] * (v - mu) * inv + lnb[l * DD + d];
        y = fmaxf(y, 0.f);
        oac[n * DD + d] = y;
        h16[n * DD + d] = (_Float16)y;
      }
    }
    __syncthreads();
  }

  // global mean pool -> f16 emb in workspace
  if (tid < DD) {
    int d = tid;
    float s = 0.f;
    for (int n = 0; n < NN; ++n) s += oac[n * DD + d];
    ws16[OFF_EMB + g * DD + d] = (_Float16)(s * (1.f / NN));
  }
}

// ---------------------------------------------------------------------------
// Kernel 3: 2-layer LSTM (interleaved per timestep) + MLP head, 1 workgroup.
// ---------------------------------------------------------------------------
__global__ __launch_bounds__(256)
void xg_lstm_head_kernel(const _Float16* __restrict__ ws16,
                         const float* __restrict__ bih0, const float* __restrict__ bhh0,
                         const float* __restrict__ bih1, const float* __restrict__ bhh1,
                         const float* __restrict__ W1, const float* __restrict__ b1,
                         const float* __restrict__ W2, const float* __restrict__ b2,
                         const float* __restrict__ W3, const float* __restrict__ b3,
                         float* __restrict__ out) {
  __shared__ _Float16 h0s[BB * SHD];
  __shared__ _Float16 h1s[BB * SHD];
  __shared__ float c0s[BB * SHD];
  __shared__ float c1s[BB * SHD];
  __shared__ float zs[BB * 512];
  __shared__ float z1s[BB * 64];
  __shared__ float z2s[BB * 32];

  const int tid  = threadIdx.x;
  const int lane = tid & 31;
  const int wave = tid >> 5;

  const _Float16* emb  = ws16 + OFF_EMB;   // [B*T][64], batch row stride T*64
  const _Float16* Wih0 = ws16 + OFF_WIH0;  // [512][64]
  const _Float16* Whh0 = ws16 + OFF_WHH0;  // [512][128]
  const _Float16* Wih1 = ws16 + OFF_WIH1;  // [512][128]
  const _Float16* Whh1 = ws16 + OFF_WHH1;  // [512][128]

  for (int i = tid; i < BB * SHD; i += 256) {
    h0s[i] = (_Float16)0.f; h1s[i] = (_Float16)0.f;
    c0s[i] = 0.f; c1s[i] = 0.f;
  }
  __syncthreads();

  for (int t = 0; t < TT; ++t) {
    // ---- layer 0: z = x_t @ Wih0.T + h0 @ Whh0.T ----
    for (int j = 0; j < 4; ++j) {
      int tn = wave * 4 + j;                       // 32 tiles of N=512
      v8f acc = {};
#pragma unroll
      for (int kc = 0; kc < 64; kc += 32) {        // x_t: [16][64], stride T*64
        v16h a = load_a_frag(emb + t * DD, TT * DD, 0, kc, lane);
        v16h b = load_b_frag(Wih0, 64, tn * 16, kc, lane);
        acc = wmma_f16(a, b, acc);
      }
#pragma unroll
      for (int kc = 0; kc < SHD; kc += 32) {
        v16h a = load_a_frag(h0s, SHD, 0, kc, lane);
        v16h b = load_b_frag(Whh0, SHD, tn * 16, kc, lane);
        acc = wmma_f16(a, b, acc);
      }
      store_c_frag(zs, 512, 0, tn * 16, lane, acc);
    }
    __syncthreads();
    for (int idx = tid; idx < BB * SHD; idx += 256) {
      int b = idx >> 7, u = idx & 127;
      float zi = zs[b * 512 + u]        + bih0[u]        + bhh0[u];
      float zf = zs[b * 512 + 128 + u]  + bih0[128 + u]  + bhh0[128 + u];
      float zg = zs[b * 512 + 256 + u]  + bih0[256 + u]  + bhh0[256 + u];
      float zo = zs[b * 512 + 384 + u]  + bih0[384 + u]  + bhh0[384 + u];
      float cc = sigm(zf) * c0s[idx] + sigm(zi) * tanhf(zg);
      c0s[idx] = cc;
      h0s[idx] = (_Float16)(sigm(zo) * tanhf(cc));
    }
    __syncthreads();
    // ---- layer 1: z = h0 @ Wih1.T + h1 @ Whh1.T ----
    for (int j = 0; j < 4; ++j) {
      int tn = wave * 4 + j;
      v8f acc = {};
#pragma unroll
      for (int kc = 0; kc < SHD; kc += 32) {
        v16h a = load_a_frag(h0s, SHD, 0, kc, lane);
        v16h b = load_b_frag(Wih1, SHD, tn * 16, kc, lane);
        acc = wmma_f16(a, b, acc);
        a = load_a_frag(h1s, SHD, 0, kc, lane);
        b = load_b_frag(Whh1, SHD, tn * 16, kc, lane);
        acc = wmma_f16(a, b, acc);
      }
      store_c_frag(zs, 512, 0, tn * 16, lane, acc);
    }
    __syncthreads();
    for (int idx = tid; idx < BB * SHD; idx += 256) {
      int b = idx >> 7, u = idx & 127;
      float zi = zs[b * 512 + u]        + bih1[u]        + bhh1[u];
      float zf = zs[b * 512 + 128 + u]  + bih1[128 + u]  + bhh1[128 + u];
      float zg = zs[b * 512 + 256 + u]  + bih1[256 + u]  + bhh1[256 + u];
      float zo = zs[b * 512 + 384 + u]  + bih1[384 + u]  + bhh1[384 + u];
      float cc = sigm(zf) * c1s[idx] + sigm(zi) * tanhf(zg);
      c1s[idx] = cc;
      h1s[idx] = (_Float16)(sigm(zo) * tanhf(cc));
    }
    __syncthreads();
  }

  // ---- MLP head: 128 -> 64 -> 32 -> 1, sigmoid ----
  for (int idx = tid; idx < BB * 64; idx += 256) {
    int b = idx >> 6, j = idx & 63;
    float s = b1[j];
    for (int k = 0; k < SHD; ++k) s += (float)h1s[b * SHD + k] * W1[j * SHD + k];
    z1s[idx] = fmaxf(s, 0.f);
  }
  __syncthreads();
  for (int idx = tid; idx < BB * 32; idx += 256) {
    int b = idx >> 5, j = idx & 31;
    float s = b2[j];
    for (int k = 0; k < 64; ++k) s += z1s[b * 64 + k] * W2[j * 64 + k];
    z2s[idx] = fmaxf(s, 0.f);
  }
  __syncthreads();
  if (tid < BB) {
    float s = b3[0];
    for (int k = 0; k < 32; ++k) s += z2s[tid * 32 + k] * W3[k];
    out[tid] = sigm(s);
  }
}

// ---------------------------------------------------------------------------
extern "C" void kernel_launch(void* const* d_in, const int* in_sizes, int n_in,
                              void* d_out, int out_size, void* d_ws, size_t ws_size,
                              hipStream_t stream) {
  const float* x     = (const float*)d_in[0];
  const int*   ei    = (const int*)d_in[1];
  const float* projW = (const float*)d_in[2];
  const float* projb = (const float*)d_in[3];
  const float* gWl   = (const float*)d_in[4];
  const float* gWr   = (const float*)d_in[5];
  const float* gatt  = (const float*)d_in[6];
  const float* gb    = (const float*)d_in[7];
  const float* lng   = (const float*)d_in[8];
  const float* lnb   = (const float*)d_in[9];
  const float* wih0  = (const float*)d_in[10];
  const float* whh0  = (const float*)d_in[11];
  const float* bih0  = (const float*)d_in[12];
  const float* bhh0  = (const float*)d_in[13];
  const float* wih1  = (const float*)d_in[14];
  const float* whh1  = (const float*)d_in[15];
  const float* bih1  = (const float*)d_in[16];
  const float* bhh1  = (const float*)d_in[17];
  const float* hW1   = (const float*)d_in[18];
  const float* hb1   = (const float*)d_in[19];
  const float* hW2   = (const float*)d_in[20];
  const float* hb2   = (const float*)d_in[21];
  const float* hW3   = (const float*)d_in[22];
  const float* hb3   = (const float*)d_in[23];

  _Float16* ws16 = (_Float16*)d_ws;
  float* out = (float*)d_out;

  xg_convert_kernel<<<(CVT_TOTAL + 255) / 256, 256, 0, stream>>>(
      gWl, gWr, wih0, whh0, wih1, whh1, ws16);
  xg_gat_kernel<<<GG, 256, 0, stream>>>(
      x, ei, projW, projb, gatt, gb, lng, lnb, ws16);
  xg_lstm_head_kernel<<<1, 256, 0, stream>>>(
      ws16, bih0, bhh0, bih1, bhh1, hW1, hb1, hW2, hb2, hW3, hb3, out);
}
